// GCNClassifier_3676492005926
// MI455X (gfx1250) — matrix operand order repaired and checked by reading, not compiled, not run on hardware
//
#include <hip/hip_runtime.h>
#include <math.h>

// ---------------- model dims ----------------
#define B_    32
#define L_    256
#define BL    8192
#define HN_   6
#define TOPK_ 64
#define NC_   3
#define RH_   300
#define HID_  300
#define DRNN  600
#define DL1   900

// padded dims (K mult of 32, N mult of 64)
#define EINP_P 384     // 360 -> 384
#define G4_P   1216    // 1200 -> 1216
#define RH_P   320     // 300 -> 320
#define HID_P  320
#define DRNN_P 640     // 600 -> 640
#define DL1_P  960     // 900 -> 960
#define DKMAX_P 160

typedef __bf16 bf16;
typedef __attribute__((ext_vector_type(8)))  __bf16 bf16x8;
typedef __attribute__((ext_vector_type(16))) __bf16 bf16x16;
typedef __attribute__((ext_vector_type(8)))  float  f32x8;
typedef __attribute__((ext_vector_type(4)))  unsigned int u32x4;
typedef __attribute__((ext_vector_type(8)))  int i32x8;
typedef __attribute__((ext_vector_type(4)))  int i32x4;

#define CAT16(lo, hi) __builtin_shufflevector(lo, hi, 0,1,2,3,4,5,6,7,8,9,10,11,12,13,14,15)

__device__ __forceinline__ f32x8 wmma_bf16(bf16x16 a, bf16x16 b, f32x8 c) {
  // (neg_a, A, neg_b, B, c_mod, C, reuse_a, reuse_b)
  return __builtin_amdgcn_wmma_f32_16x16x32_bf16(false, a, false, b, (short)0, c, false, false);
}

// ---------------- Tensor Data Mover: 2D bf16 tile -> LDS ----------------
// Loads tile_d1 rows x tile_d0 elements (2B each) from a row-major tensor with
// row stride 'stride' (elements) into LDS at lds_off. LDS padding: after every
// 16 DWORDs (one 64B tile row) insert 4 DWORDs (16B) -> LDS row stride 80B,
// which matches the AS_STR/BS_STR = 40-half layout used by the WMMA readers.
__device__ __forceinline__ void tdm_load_2d(unsigned lds_off, const void* gaddr,
                                            unsigned tile_d0, unsigned tile_d1,
                                            unsigned long stride)
{
  unsigned long ga = (unsigned long)gaddr;
  u32x4 g0;
  g0[0] = 1u;                                               // count=1 (valid user D#)
  g0[1] = lds_off;                                          // lds_addr (bytes)
  g0[2] = (unsigned)(ga & 0xffffffffu);                     // global_addr[31:0]
  g0[3] = (unsigned)((ga >> 32) & 0x01ffffffu) | 0x80000000u; // global_addr[56:32] | type=2
  i32x8 g1;
  g1[0] = (1 << 16) | (1 << 20) | (3 << 22) | (3 << 25);    // data_size=2B, pad_en, intv=16dw, amt=4dw
  g1[1] = (int)((tile_d0 & 0xffffu) << 16);                 // tensor_dim0[15:0]
  g1[2] = (int)((tile_d0 >> 16) | ((tile_d1 & 0xffffu) << 16)); // tensor_dim0[31:16] | tensor_dim1[15:0]
  g1[3] = (int)((tile_d1 >> 16) | (tile_d0 << 16));         // tensor_dim1[31:16] | tile_dim0
  g1[4] = (int)(tile_d1 & 0xffffu);                         // tile_dim1 (tile_dim2=0)
  g1[5] = (int)(stride & 0xffffffffu);                      // tensor_dim0_stride[31:0]
  g1[6] = (int)((stride >> 32) & 0xffffu);                  // tensor_dim0_stride[47:32]
  g1[7] = 0;
  i32x4 z4 = {0, 0, 0, 0};
#if defined(__clang_major__) && (__clang_major__ >= 23)
  i32x8 z8 = {0, 0, 0, 0, 0, 0, 0, 0};
  __builtin_amdgcn_tensor_load_to_lds(g0, g1, z4, z4, z8, 0);
#else
  __builtin_amdgcn_tensor_load_to_lds(g0, g1, z4, z4, 0);
#endif
}

// ---------------- generic WMMA GEMM ----------------
// C[M,N] (f32) = A[M,K](bf16) * op(B) + bias, op(B)=B^T if BT (B is [N,K]) else B is [K,N].
// Block tile 128x64, 8 waves as 4(M)x2(N), each wave 32x32 via 2x2 wmma.
// Double-buffered LDS: TDM DMA for slab k+1 overlaps WMMA on slab k.
#define AS_STR 40
#define BS_STR 40

template <bool BT>
__global__ __launch_bounds__(256)
void k_gemm(const bf16* __restrict__ A, const bf16* __restrict__ Bm,
            float* __restrict__ C, const float* __restrict__ bias, int biasN,
            int M, int N, int K, int lda, int ldb, int ldc,
            long sA, long sB, long sC)
{
  __shared__ alignas(16) bf16 As[2][128 * AS_STR];
  __shared__ alignas(16) bf16 Bs[2][64 * BS_STR];
  int z = blockIdx.z;
  A  += (long)z * sA;
  Bm += (long)z * sB;
  C  += (long)z * sC;
  int m0 = blockIdx.x * 128, n0 = blockIdx.y * 64;
  int tid = threadIdx.x, lane = tid & 31, wid = tid >> 5;
  int wm = wid & 3, wn = wid >> 2;
  int lh = lane & 15, hi = lane >> 4;
  unsigned as_off[2] = { (unsigned)(size_t)(void*)As[0], (unsigned)(size_t)(void*)As[1] };
  unsigned bs_off[2] = { (unsigned)(size_t)(void*)Bs[0], (unsigned)(size_t)(void*)Bs[1] };
  f32x8 acc[2][2] = {};

  int nk = K >> 5;
  // prologue: DMA slab 0 into buffer 0
  if (wid == 0) {
    tdm_load_2d(as_off[0], A + (long)m0 * lda, 32, 128, (unsigned long)lda);
    if (BT) tdm_load_2d(bs_off[0], Bm + (long)n0 * ldb, 32, 64, (unsigned long)ldb);
  }

  for (int ki = 0; ki < nk; ++ki) {
    int cur = ki & 1, nxt = cur ^ 1;
    int k0 = ki << 5;
    if (wid == 0) __builtin_amdgcn_s_wait_tensorcnt(0);   // slab ki landed in LDS
    if (!BT) {
      // manual transpose-stage of B[K,N] slab ki into Bs[cur][n][k]
      int kk = tid >> 3, nc = (tid & 7) * 8;
      bf16x8 v = {};
      if (n0 + nc < N) {
        const bf16* p = Bm + (long)(k0 + kk) * ldb + n0 + nc;
        v = *(const bf16x8*)p;
        if (ki + 1 < nk) __builtin_prefetch(p + 32 * ldb, 0, 1);
      }
#pragma unroll
      for (int e = 0; e < 8; ++e) Bs[cur][(nc + e) * BS_STR + kk] = v[e];
    }
    __syncthreads();
    // kick DMA for slab ki+1 while everyone computes slab ki
    if (wid == 0 && ki + 1 < nk) {
      tdm_load_2d(as_off[nxt], A + (long)m0 * lda + k0 + 32, 32, 128, (unsigned long)lda);
      if (BT) tdm_load_2d(bs_off[nxt], Bm + (long)n0 * ldb + k0 + 32, 32, 64, (unsigned long)ldb);
    }

    const bf16* Asb = As[cur];
    const bf16* Bsb = Bs[cur];
#pragma unroll
    for (int mt = 0; mt < 2; ++mt) {
      const bf16* ar = Asb + (wm * 32 + mt * 16 + lh) * AS_STR;
      bf16x8 a_lo = *(const bf16x8*)(ar + hi * 8);        // K {0..7 | 8..15}
      bf16x8 a_hi = *(const bf16x8*)(ar + 16 + hi * 8);   // K {16..23 | 24..31}
      bf16x16 a = CAT16(a_lo, a_hi);
#pragma unroll
      for (int nt = 0; nt < 2; ++nt) {
        const bf16* br = Bsb + (wn * 32 + nt * 16 + lh) * BS_STR + hi * 8;
        bf16x8 b_lo = *(const bf16x8*)(br);
        bf16x8 b_hi = *(const bf16x8*)(br + 16);
        bf16x16 b = CAT16(b_lo, b_hi);
        acc[mt][nt] = wmma_bf16(a, b, acc[mt][nt]);
      }
    }
    // no trailing barrier: next iteration's top barrier orders buffer reuse
  }

  // ---- store C (D layout: elem r -> row base+hi*8+r, col = lane&15) ----
#pragma unroll
  for (int mt = 0; mt < 2; ++mt)
#pragma unroll
    for (int nt = 0; nt < 2; ++nt) {
      int cr = m0 + wm * 32 + mt * 16 + hi * 8;
      int cn = n0 + wn * 32 + nt * 16 + lh;
      if (cn < N) {
        float bv = (bias && cn < biasN) ? bias[cn] : 0.f;
#pragma unroll
        for (int r = 0; r < 8; ++r) {
          int rr = cr + r;
          if (rr < M) C[(long)rr * ldc + cn] = acc[mt][nt][r] + bv;
        }
      }
    }
}

// ---------------- conversion / padding ----------------
__global__ void k_pad_cvt(bf16* __restrict__ dst, const float* __restrict__ src,
                          int R, int C, int Cp, int srcld, long total)
{
  for (long idx = (long)blockIdx.x * blockDim.x + threadIdx.x; idx < total;
       idx += (long)gridDim.x * blockDim.x) {
    long r = idx / Cp; int c = (int)(idx - r * Cp);
    float v = (r < R && c < C) ? src[r * (long)srcld + c] : 0.f;
    dst[idx] = (bf16)v;
  }
}

__global__ void k_bias_sum(float* __restrict__ dst, const float* __restrict__ a,
                           const float* __restrict__ b, int n, int np)
{
  int i = blockIdx.x * blockDim.x + threadIdx.x;
  if (i < np) dst[i] = (i < n) ? a[i] + b[i] : 0.f;
}

// ---------------- embedding gather ----------------
__global__ void k_embed(bf16* __restrict__ dst, const int* __restrict__ tok,
                        const int* __restrict__ pos, const int* __restrict__ post,
                        const float* __restrict__ eW, const float* __restrict__ pW,
                        const float* __restrict__ sW)
{
  long idx = (long)blockIdx.x * blockDim.x + threadIdx.x;
  long total = (long)BL * EINP_P;
  if (idx >= total) return;
  long r = idx / EINP_P; int c = (int)(idx - r * EINP_P);
  float v = 0.f;
  if (c < 300)       v = eW[(long)tok[r] * 300 + c];
  else if (c < 330)  v = pW[(long)pos[r] * 30 + (c - 300)];
  else if (c < 360)  v = sW[(long)post[r] * 30 + (c - 330)];
  dst[idx] = (bf16)v;
}

// ---------------- LSTM: persistent 32-wave workgroup ----------------
__global__ __launch_bounds__(1024)
void k_lstm(const float* __restrict__ xg, const bf16* __restrict__ wh,
            float* __restrict__ g_ws, float* __restrict__ c_ws,
            float* __restrict__ rnn, int backward)
{
  __shared__ alignas(16) bf16 hbf[32 * RH_P];
  int tid = threadIdx.x;
  for (int i = tid; i < 32 * RH_P; i += 1024) hbf[i] = (bf16)0.f;
  for (int i = tid; i < 32 * RH_; i += 1024) c_ws[i] = 0.f;
  __syncthreads();
  int lane = tid & 31, wid = tid >> 5;
  int lh = lane & 15, hi = lane >> 4;

  for (int s = 0; s < L_; ++s) {
    int t = backward ? (L_ - 1 - s) : s;
    // g[32,1216] = hbf @ wh^T + xg[:,t,:]
    for (int tile = wid; tile < 2 * (G4_P / 16); tile += 32) {
      int mt = tile & 1, nt = tile >> 1;
      f32x8 acc = {};
      const bf16* ar = hbf + (mt * 16 + lh) * RH_P;
      const bf16* brow = wh + (long)(nt * 16 + lh) * RH_P;
      // software-pipelined B loads: fetch slab k+1 before the WMMA on slab k
      bf16x8 b_lo = *(const bf16x8*)(brow + hi * 8);
      bf16x8 b_hi = *(const bf16x8*)(brow + 16 + hi * 8);
      for (int k0 = 0; k0 < RH_P; k0 += 32) {
        bf16x16 b = CAT16(b_lo, b_hi);
        if (k0 + 32 < RH_P) {
          const bf16* bp = brow + k0 + 32 + hi * 8;
          b_lo = *(const bf16x8*)(bp);
          b_hi = *(const bf16x8*)(bp + 16);
        }
        bf16x8 a_lo = *(const bf16x8*)(ar + k0 + hi * 8);
        bf16x8 a_hi = *(const bf16x8*)(ar + k0 + 16 + hi * 8);
        bf16x16 a = CAT16(a_lo, a_hi);
        acc = wmma_bf16(a, b, acc);
      }
      int cr = mt * 16 + hi * 8, cn = nt * 16 + lh;
#pragma unroll
      for (int r = 0; r < 8; ++r) {
        int m = cr + r;
        g_ws[m * G4_P + cn] = acc[r] + xg[((long)m * L_ + t) * G4_P + cn];
      }
    }
    __threadfence();
    __syncthreads();
    // gates + state update
    for (int j = tid; j < 32 * RH_; j += 1024) {
      int m = j / RH_, u = j - m * RH_;
      const float* gr = g_ws + m * G4_P;
      float ig = gr[u], fg = gr[RH_ + u], gg = gr[2 * RH_ + u], og = gr[3 * RH_ + u];
      float si = 1.f / (1.f + __expf(-ig));
      float sf = 1.f / (1.f + __expf(-fg));
      float so = 1.f / (1.f + __expf(-og));
      float c = sf * c_ws[j] + si * tanhf(gg);
      float h = so * tanhf(c);
      c_ws[j] = c;
      rnn[((long)m * L_ + t) * DRNN_P + (backward ? RH_ : 0) + u] = h;
      hbf[m * RH_P + u] = (bf16)h;
    }
    __syncthreads();
  }
}

// ---------------- attention helpers ----------------
__global__ void k_head_gather(bf16* __restrict__ dst, const float* __restrict__ src,
                              int Dp, int dk, int dkp, long total)
{
  long idx = (long)blockIdx.x * blockDim.x + threadIdx.x;
  if (idx >= total) return;
  long t = idx; int kk = (int)(t % dkp); t /= dkp;
  int i = (int)(t % L_); t /= L_;
  int h = (int)(t % HN_); int b = (int)(t / HN_);
  float v = (kk < dk) ? src[((long)b * L_ + i) * Dp + h * dk + kk] : 0.f;
  dst[idx] = (bf16)v;
}

__global__ __launch_bounds__(256)
void k_softmax(float* __restrict__ s, const float* __restrict__ score, float inv_s)
{
  __shared__ float red[256];
  long row = blockIdx.x;
  int b = (int)(row / (HN_ * L_));
  int i = (int)(row % L_);
  float* sr = s + row * L_;
  const float* scr = score + ((long)b * L_ + i) * L_;
  int j = threadIdx.x;
  bool masked = (scr[j] == 0.f);
  float v = masked ? -1e9f : sr[j] * inv_s;
  red[j] = v; __syncthreads();
  for (int st = 128; st > 0; st >>= 1) { if (j < st) red[j] = fmaxf(red[j], red[j + st]); __syncthreads(); }
  float mx = red[0]; __syncthreads();
  float e = __expf(v - mx);
  red[j] = e; __syncthreads();
  for (int st = 128; st > 0; st >>= 1) { if (j < st) red[j] += red[j + st]; __syncthreads(); }
  float sum = red[0];
  float brow = (scr[0] == 0.f) ? 0.f : 1.f;
  sr[j] = e / sum * brow;
}

__global__ void k_headsum(float* __restrict__ att, const float* __restrict__ s)
{
  long idx = (long)blockIdx.x * blockDim.x + threadIdx.x;
  long total = (long)B_ * L_ * L_;
  if (idx >= total) return;
  long b = idx / ((long)L_ * L_); long r = idx % ((long)L_ * L_);
  float a = 0.f;
  for (int h = 0; h < HN_; ++h) a += s[((long)(b * HN_ + h)) * L_ * L_ + r];
  att[idx] = a;
}

// top-k threshold via binary search on value
__global__ __launch_bounds__(1024)
void k_topk(const float* __restrict__ att, float* __restrict__ thr)
{
  __shared__ int red[1024];
  int b = blockIdx.x;
  const float* f = att + (long)b * L_ * L_;
  float lo = 0.f, hiv = 8.f;
  for (int it = 0; it < 28; ++it) {
    float mid = 0.5f * (lo + hiv);
    int c = 0;
    for (int i = threadIdx.x; i < L_ * L_; i += 1024) c += (f[i] >= mid) ? 1 : 0;
    red[threadIdx.x] = c; __syncthreads();
    for (int st = 512; st > 0; st >>= 1) { if (threadIdx.x < st) red[threadIdx.x] += red[threadIdx.x + st]; __syncthreads(); }
    int total = red[0];
    __syncthreads();
    if (total >= TOPK_) lo = mid; else hiv = mid;
  }
  if (threadIdx.x == 0) thr[b] = lo;
}

__global__ void k_select(float* __restrict__ out, bf16* __restrict__ out_bf,
                         const float* __restrict__ a, const float* __restrict__ thr)
{
  long idx = (long)blockIdx.x * blockDim.x + threadIdx.x;
  long total = (long)B_ * L_ * L_;
  if (idx >= total) return;
  int b = (int)(idx / ((long)L_ * L_));
  long r = idx % ((long)L_ * L_);
  int i = (int)(r / L_), j = (int)(r % L_);
  float t = thr[b];
  float aij = a[idx];
  float aji = a[(long)b * L_ * L_ + (long)j * L_ + i];
  float sv = (i == j) ? 1.f : ((aij >= t ? 1.f : 0.f) + (aji >= t ? 1.f : 0.f));
  float v = sv * aij;
  out[idx] = v;
  out_bf[idx] = (bf16)v;
}

__global__ void k_denom(float* __restrict__ den, const float* __restrict__ adjm)
{
  int b = blockIdx.x, i = threadIdx.x;
  const float* row = adjm + ((long)b * L_ + i) * L_;
  float s = 1.f;
  for (int j = 0; j < L_; ++j) s += row[j];
  den[b * L_ + i] = s;
}

// ---------------- GCN pieces ----------------
__global__ void k_gcn_combine(float* __restrict__ out, const float* __restrict__ t1,
                              const float* __restrict__ t2, const float* __restrict__ den,
                              long total)
{
  long idx = (long)blockIdx.x * blockDim.x + threadIdx.x;
  if (idx >= total) return;
  long r = idx / HID_P;
  out[idx] = fmaxf(t1[idx] / den[r], 0.f) + t2[idx];
}

__global__ void k_build_inp(bf16* __restrict__ inp, const float* __restrict__ rnn,
                            const float* __restrict__ out0)
{
  long idx = (long)blockIdx.x * blockDim.x + threadIdx.x;
  long total = (long)BL * DL1_P;
  if (idx >= total) return;
  long r = idx / DL1_P; int c = (int)(idx - r * DL1_P);
  float v = 0.f;
  if (c < DRNN)      v = rnn[r * DRNN_P + c];
  else if (c < DL1)  v = out0[r * HID_P + (c - DRNN)];
  inp[idx] = (bf16)v;
}

// ---------------- pooling / heads ----------------
__global__ void k_means(float* __restrict__ means, const float* __restrict__ hsy,
                        const float* __restrict__ hse, const float* __restrict__ hcsy,
                        const float* __restrict__ hcse, const float* __restrict__ mask,
                        const float* __restrict__ hw)
{
  int b = blockIdx.x;
  float wn = 0.f;
  for (int l = 0; l < L_; ++l) wn += mask[b * L_ + l];
  float w0 = hw[0] * 0.5f, w1 = hw[1] * 0.5f;
  for (int d = threadIdx.x; d < HID_; d += blockDim.x) {
    float s1 = 0.f, s2 = 0.f, s3 = 0.f;
    for (int l = 0; l < L_; ++l) {
      float m = mask[b * L_ + l];
      if (m != 0.f) {
        long r = ((long)b * L_ + l) * HID_P;
        s1 += hsy[r + d] * m;
        s2 += hse[r + d] * m;
        s3 += (w0 * hcsy[r + d] + w1 * hcse[r + d]) * m;
      }
    }
    means[b * 900 + d] = s1 / wn;
    means[b * 900 + 300 + d] = s2 / wn;
    means[b * 900 + 600 + d] = s3 / wn;
  }
}

__global__ void k_lin(float* __restrict__ out_d, float* __restrict__ out_ws,
                      const float* __restrict__ means, const float* __restrict__ W,
                      const float* __restrict__ bias)
{
  int b = blockIdx.x;
  for (int o = threadIdx.x; o < HID_; o += blockDim.x) {
    float a = bias[o];
    const float* mr = means + (long)b * 900;
    const float* wr = W + (long)o * (3 * HID_);
    for (int j = 0; j < 3 * HID_; ++j) a += mr[j] * wr[j];
    a = fmaxf(a, 0.f);
    out_ws[b * HID_ + o] = a;
    out_d[b * HID_ + o] = a;
  }
}

__global__ void k_cls(float* __restrict__ logits, const float* __restrict__ outs,
                      const float* __restrict__ W, const float* __restrict__ bias)
{
  int idx = blockIdx.x * blockDim.x + threadIdx.x;
  if (idx >= B_ * NC_) return;
  int b = idx / NC_, o = idx % NC_;
  float a = bias[o];
  for (int j = 0; j < HID_; ++j) a += outs[b * HID_ + j] * W[o * HID_ + j];
  logits[idx] = a;
}

__global__ void k_strip(float* __restrict__ dst, const float* __restrict__ src)
{
  long idx = (long)blockIdx.x * blockDim.x + threadIdx.x;
  long total = (long)BL * HID_;
  if (idx >= total) return;
  long r = idx / HID_; int c = (int)(idx - r * HID_);
  dst[idx] = src[r * HID_P + c];
}

// ---------------- host orchestration ----------------
static inline dim3 g1d(long n) { return dim3((unsigned)((n + 255) / 256)); }

extern "C" void kernel_launch(void* const* d_in, const int* in_sizes, int n_in,
                              void* d_out, int out_size, void* d_ws, size_t ws_size,
                              hipStream_t stream)
{
  (void)in_sizes; (void)n_in; (void)out_size; (void)ws_size;
  const int*   tok   = (const int*)d_in[0];
  const int*   pos   = (const int*)d_in[1];
  const int*   post  = (const int*)d_in[2];
  const float* mask  = (const float*)d_in[3];
  const float* adj   = (const float*)d_in[5];
  const float* embW  = (const float*)d_in[6];
  const float* posW  = (const float*)d_in[7];
  const float* postW = (const float*)d_in[8];
  const float* wi_f  = (const float*)d_in[9];
  const float* wh_f  = (const float*)d_in[10];
  const float* bi_f  = (const float*)d_in[11];
  const float* bh_f  = (const float*)d_in[12];
  const float* wi_b  = (const float*)d_in[13];
  const float* wh_b  = (const float*)d_in[14];
  const float* bi_b  = (const float*)d_in[15];
  const float* bh_b  = (const float*)d_in[16];
  const float* qW    = (const float*)d_in[17];
  const float* qb    = (const float*)d_in[18];
  const float* kW    = (const float*)d_in[19];
  const float* kb    = (const float*)d_in[20];
  const float* hwv   = (const float*)d_in[21];
  const float* linW  = (const float*)d_in[22];
  const float* linb  = (const float*)d_in[23];
  const float* clsW  = (const float*)d_in[24];
  const float* clsb  = (const float*)d_in[25];
  const float* gp[3][8];
  for (int n = 0; n < 3; ++n)
    for (int k = 0; k < 8; ++k) gp[n][k] = (const float*)d_in[26 + n * 8 + k];

  // ---- workspace carve-out ----
  uintptr_t base = (uintptr_t)d_ws; size_t off = 0;
  auto alloc = [&](size_t bytes) -> void* {
    off = (off + 255) & ~(size_t)255;
    void* p = (void*)(base + off);
    off += bytes;
    return p;
  };
  bf16* wiF = (bf16*)alloc((size_t)G4_P * EINP_P * 2);
  bf16* whF = (bf16*)alloc((size_t)G4_P * RH_P * 2);
  bf16* wiB = (bf16*)alloc((size_t)G4_P * EINP_P * 2);
  bf16* whB = (bf16*)alloc((size_t)G4_P * RH_P * 2);
  bf16* qWb = (bf16*)alloc((size_t)DRNN_P * DRNN_P * 2);
  bf16* kWb = (bf16*)alloc((size_t)DRNN_P * DRNN_P * 2);
  bf16 *W0b[3], *W1b[3], *aqWb[3] = {}, *akWb[3] = {};
  for (int n = 0; n < 3; ++n) {
    W0b[n] = (bf16*)alloc((size_t)HID_P * DRNN_P * 2);
    W1b[n] = (bf16*)alloc((size_t)HID_P * DL1_P * 2);
  }
  for (int n = 1; n < 3; ++n) {
    aqWb[n] = (bf16*)alloc((size_t)DL1_P * DL1_P * 2);
    akWb[n] = (bf16*)alloc((size_t)DL1_P * DL1_P * 2);
  }
  float* bsF = (float*)alloc(G4_P * 4);
  float* bsB = (float*)alloc(G4_P * 4);
  bf16*  embs = (bf16*)alloc((size_t)BL * EINP_P * 2);
  float* xgF  = (float*)alloc((size_t)BL * G4_P * 4);
  float* xgB  = (float*)alloc((size_t)BL * G4_P * 4);
  float* gws0 = (float*)alloc(32 * G4_P * 4);
  float* cws0 = (float*)alloc(32 * RH_ * 4);
  float* gws1 = (float*)alloc(32 * G4_P * 4);
  float* cws1 = (float*)alloc(32 * RH_ * 4);
  float* rnnF = (float*)alloc((size_t)BL * DRNN_P * 4);
  bf16*  rnnB = (bf16*)alloc((size_t)BL * DRNN_P * 2);
  float* score = (float*)alloc((size_t)B_ * L_ * L_ * 4);
  float* qf = xgF;  // alias: xg no longer needed after LSTM
  float* kf = xgB;
  bf16*  qh = (bf16*)alloc((size_t)B_ * HN_ * L_ * DKMAX_P * 2);
  bf16*  kh = (bf16*)alloc((size_t)B_ * HN_ * L_ * DKMAX_P * 2);
  float* sws = (float*)alloc((size_t)B_ * HN_ * L_ * L_ * 4);
  float* att = (float*)alloc((size_t)B_ * L_ * L_ * 4);
  float* attadj = (float*)alloc((size_t)B_ * L_ * L_ * 4);
  bf16*  attadj_b = (bf16*)alloc((size_t)B_ * L_ * L_ * 2);
  bf16*  adj_b = (bf16*)alloc((size_t)B_ * L_ * L_ * 2);
  float* a2 = (float*)alloc((size_t)B_ * L_ * L_ * 4);
  bf16*  a2_b = (bf16*)alloc((size_t)B_ * L_ * L_ * 2);
  float* denA  = (float*)alloc(B_ * L_ * 4);
  float* denAA = (float*)alloc(B_ * L_ * 4);
  float* den2  = (float*)alloc(B_ * L_ * 4);
  float* thr   = (float*)alloc(B_ * 4);
  float* Ax  = (float*)alloc((size_t)BL * DL1_P * 4);
  bf16*  Axb = (bf16*)alloc((size_t)BL * DL1_P * 2);
  float* t1  = (float*)alloc((size_t)BL * HID_P * 4);
  float* t2  = (float*)alloc((size_t)BL * HID_P * 4);
  float* out0 = (float*)alloc((size_t)BL * HID_P * 4);
  bf16*  inpb = (bf16*)alloc((size_t)BL * DL1_P * 2);
  float* hsy  = (float*)alloc((size_t)BL * HID_P * 4);
  float* hse  = (float*)alloc((size_t)BL * HID_P * 4);
  float* hcsy = (float*)alloc((size_t)BL * HID_P * 4);
  float* hcse = (float*)alloc((size_t)BL * HID_P * 4);
  float* means = (float*)alloc(B_ * 900 * 4);
  float* outs  = (float*)alloc(B_ * HID_ * 4);

  float* o = (float*)d_out;
  float* o_logits = o;
  float* o_outputs = o + B_ * NC_;
  float* o_h[4];
  o_h[0] = o_outputs + B_ * HID_;
  for (int i = 1; i < 4; ++i) o_h[i] = o_h[i - 1] + (long)BL * HID_;

  // ---- helpers ----
  auto cvt = [&](bf16* dst, const float* src, int R, int C, int Rp, int Cp, int srcld) {
    long tot = (long)Rp * Cp;
    k_pad_cvt<<<g1d(tot), 256, 0, stream>>>(dst, src, R, C, Cp, srcld, tot);
  };
  auto gemm_nt = [&](const bf16* A, const bf16* Bm, float* C, const float* bias, int biasN,
                     int M, int N, int K, int lda, int ldb, int ldc,
                     long sA, long sB, long sC, int nb) {
    dim3 g((M + 127) / 128, (N + 63) / 64, nb);
    k_gemm<true><<<g, 256, 0, stream>>>(A, Bm, C, bias, biasN, M, N, K, lda, ldb, ldc, sA, sB, sC);
  };
  auto gemm_nn = [&](const bf16* A, const bf16* Bm, float* C,
                     int M, int N, int K, int lda, int ldb, int ldc,
                     long sA, long sB, long sC, int nb) {
    dim3 g((M + 127) / 128, (N + 63) / 64, nb);
    k_gemm<false><<<g, 256, 0, stream>>>(A, Bm, C, (const float*)nullptr, 0, M, N, K, lda, ldb, ldc, sA, sB, sC);
  };
  auto mha = [&](const bf16* x_bf, int Dp, int D, const float* bq, const float* bk,
                 const bf16* qWbf, const bf16* kWbf, float* a_out) {
    int dk = D / HN_;
    int dkp = ((dk + 31) / 32) * 32;
    gemm_nt(x_bf, qWbf, qf, bq, D, BL, Dp, Dp, Dp, Dp, Dp, 0, 0, 0, 1);
    gemm_nt(x_bf, kWbf, kf, bk, D, BL, Dp, Dp, Dp, Dp, Dp, 0, 0, 0, 1);
    long tg = (long)B_ * HN_ * L_ * dkp;
    k_head_gather<<<g1d(tg), 256, 0, stream>>>(qh, qf, Dp, dk, dkp, tg);
    k_head_gather<<<g1d(tg), 256, 0, stream>>>(kh, kf, Dp, dk, dkp, tg);
    gemm_nt(qh, kh, sws, nullptr, 0, L_, L_, dkp, dkp, dkp, L_,
            (long)L_ * dkp, (long)L_ * dkp, (long)L_ * L_, B_ * HN_);
    k_softmax<<<dim3(B_ * HN_ * L_), L_, 0, stream>>>(sws, score, 1.f / sqrtf((float)dk));
    k_headsum<<<g1d((long)B_ * L_ * L_), 256, 0, stream>>>(a_out, sws);
  };
  auto gcn_layer = [&](const bf16* adjm, const float* den, const bf16* xb, int Xp,
                       const bf16* Wb, const float* bias, float* outp) {
    gemm_nn(adjm, xb, Ax, L_, Xp, L_, L_, Xp, Xp,
            (long)L_ * L_, (long)L_ * Xp, (long)L_ * Xp, B_);
    cvt(Axb, Ax, BL, Xp, BL, Xp, Xp);
    gemm_nt(Axb, Wb, t1, bias, HID_, BL, HID_P, Xp, Xp, Xp, HID_P, 0, 0, 0, 1);
    gemm_nt(xb, Wb, t2, bias, HID_, BL, HID_P, Xp, Xp, Xp, HID_P, 0, 0, 0, 1);
    k_gcn_combine<<<g1d((long)BL * HID_P), 256, 0, stream>>>(outp, t1, t2, den, (long)BL * HID_P);
  };
  auto gcn = [&](const bf16* adjm, const float* den_adjm, int n, bool semantic, float* hout) {
    gcn_layer(adjm, den_adjm, rnnB, DRNN_P, W0b[n], gp[n][1], out0);
    k_build_inp<<<g1d((long)BL * DL1_P), 256, 0, stream>>>(inpb, rnnF, out0);
    const bf16* adj2 = adjm;
    const float* d2 = den_adjm;
    if (semantic) {
      mha(inpb, DL1_P, DL1, gp[n][5], gp[n][7], aqWb[n], akWb[n], att);
      k_topk<<<dim3(B_), 1024, 0, stream>>>(att, thr);
      k_select<<<g1d((long)B_ * L_ * L_), 256, 0, stream>>>(a2, a2_b, att, thr);
      k_denom<<<dim3(B_), L_, 0, stream>>>(den2, a2);
      adj2 = a2_b; d2 = den2;
    }
    gcn_layer(adj2, d2, inpb, DL1_P, W1b[n], gp[n][3], hout);
  };

  // ---- 1) weight conversion to padded bf16 ----
  cvt(wiF, wi_f, 1200, 360, G4_P, EINP_P, 360);
  cvt(whF, wh_f, 1200, 300, G4_P, RH_P, 300);
  cvt(wiB, wi_b, 1200, 360, G4_P, EINP_P, 360);
  cvt(whB, wh_b, 1200, 300, G4_P, RH_P, 300);
  cvt(qWb, qW, DRNN, DRNN, DRNN_P, DRNN_P, DRNN);
  cvt(kWb, kW, DRNN, DRNN, DRNN_P, DRNN_P, DRNN);
  for (int n = 0; n < 3; ++n) {
    cvt(W0b[n], gp[n][0], HID_, DRNN, HID_P, DRNN_P, DRNN);
    cvt(W1b[n], gp[n][2], HID_, DL1, HID_P, DL1_P, DL1);
  }
  for (int n = 1; n < 3; ++n) {
    cvt(aqWb[n], gp[n][4], DL1, DL1, DL1_P, DL1_P, DL1);
    cvt(akWb[n], gp[n][6], DL1, DL1, DL1_P, DL1_P, DL1);
  }
  k_bias_sum<<<g1d(G4_P), 256, 0, stream>>>(bsF, bi_f, bh_f, 1200, G4_P);
  k_bias_sum<<<g1d(G4_P), 256, 0, stream>>>(bsB, bi_b, bh_b, 1200, G4_P);

  // ---- 2) embeddings + LSTM input gates ----
  k_embed<<<g1d((long)BL * EINP_P), 256, 0, stream>>>(embs, tok, pos, post, embW, posW, postW);
  gemm_nt(embs, wiF, xgF, bsF, G4_P, BL, G4_P, EINP_P, EINP_P, EINP_P, G4_P, 0, 0, 0, 1);
  gemm_nt(embs, wiB, xgB, bsB, G4_P, BL, G4_P, EINP_P, EINP_P, EINP_P, G4_P, 0, 0, 0, 1);

  // ---- 3) bidirectional LSTM (persistent WMMA workgroup) ----
  k_lstm<<<dim3(1), 1024, 0, stream>>>(xgF, whF, gws0, cws0, rnnF, 0);
  k_lstm<<<dim3(1), 1024, 0, stream>>>(xgB, whB, gws1, cws1, rnnF, 1);
  cvt(rnnB, rnnF, BL, DRNN, BL, DRNN_P, DRNN_P);

  // ---- 4) self score + mask source ----
  gemm_nt(rnnB, rnnB, score, nullptr, 0, L_, L_, DRNN_P, DRNN_P, DRNN_P, L_,
          (long)L_ * DRNN_P, (long)L_ * DRNN_P, (long)L_ * L_, B_);

  // ---- 5) main attention -> att_adj ----
  mha(rnnB, DRNN_P, DRNN, qb, kb, qWb, kWb, att);
  k_topk<<<dim3(B_), 1024, 0, stream>>>(att, thr);
  k_select<<<g1d((long)B_ * L_ * L_), 256, 0, stream>>>(attadj, attadj_b, att, thr);
  k_denom<<<dim3(B_), L_, 0, stream>>>(denAA, attadj);

  // syntactic adjacency (given) -> bf16 + denom
  cvt(adj_b, adj, BL, L_, BL, L_, L_);
  k_denom<<<dim3(B_), L_, 0, stream>>>(denA, adj);

  // ---- 6) four GCN stacks ----
  gcn(adj_b, denA, 0, false, hsy);
  gcn(attadj_b, denAA, 1, true, hse);
  gcn(adj_b, denA, 2, false, hcsy);
  gcn(attadj_b, denAA, 2, true, hcse);

  // ---- 7) pooling + heads ----
  k_means<<<dim3(B_), 256, 0, stream>>>(means, hsy, hse, hcsy, hcse, mask, hwv);
  k_lin<<<dim3(B_), 256, 0, stream>>>(o_outputs, outs, means, linW, linb);
  k_cls<<<g1d(B_ * NC_), 256, 0, stream>>>(o_logits, outs, clsW, clsb);

  // ---- 8) strip padding into d_out ----
  k_strip<<<g1d((long)BL * HID_), 256, 0, stream>>>(o_h[0], hsy);
  k_strip<<<g1d((long)BL * HID_), 256, 0, stream>>>(o_h[1], hse);
  k_strip<<<g1d((long)BL * HID_), 256, 0, stream>>>(o_h[2], hcsy);
  k_strip<<<g1d((long)BL * HID_), 256, 0, stream>>>(o_h[3], hcse);
}
